// RNN2RNN_45509473468546
// MI455X (gfx1250) — matrix operand order repaired and checked by reading, not compiled, not run on hardware
//
#include <hip/hip_runtime.h>
#include <hip/hip_bf16.h>

// ---------------------------------------------------------------------------
// Seq2Seq GRU (enc 2-layer -> dec 2-layer -> vocab projection) for gfx1250.
// - All matmuls on v_wmma_f32_16x16x32_bf16.
// - GEMM tiles staged with global_load_async_to_lds_b128 (ASYNCcnt) into
//   double-buffered LDS; s_wait_asynccnt + barrier handoff.
// - Weights converted fp32->bf16 once; recurrence keeps its Whh slice
//   resident in CDNA5's 320KB LDS across all timesteps.
// S=64 T=48 B=32 V=32000 E=H=512, 3H=1536.
// ---------------------------------------------------------------------------

typedef __attribute__((ext_vector_type(16))) __bf16 v16bf;
typedef __attribute__((ext_vector_type(8)))  __bf16 v8bf;
typedef __attribute__((ext_vector_type(8)))  float  v8f;

static __device__ __forceinline__ v16bf cat16(v8bf a, v8bf b) {
  return __builtin_shufflevector(a, b, 0,1,2,3,4,5,6,7,8,9,10,11,12,13,14,15);
}
static __device__ __forceinline__ v8f wmma_bf16(v16bf a, v16bf b, v8f c) {
  // 8 args: (neg_a, A, neg_b, B, c_mod, C, reuse_a, reuse_b)
  return __builtin_amdgcn_wmma_f32_16x16x32_bf16(false, a, false, b, (short)0, c,
                                                 false, false);
}
static __device__ __forceinline__ float sigmoidf_(float x) {
  return 1.0f / (1.0f + expf(-x));
}

// generic shared pointer -> LDS byte offset (addr[31:0] per flat aperture map)
static __device__ __forceinline__ unsigned lds_off(const void* p) {
  return (unsigned)(unsigned long long)p;
}
// per-lane async copy of 16 bytes: global memory -> LDS (ASYNCcnt-tracked)
static __device__ __forceinline__ void async_to_lds_b128(unsigned lds,
                                                         const void* g) {
  asm volatile("global_load_async_to_lds_b128 %0, %1, off"
               :: "v"(lds), "v"(g) : "memory");
}
static __device__ __forceinline__ void wait_async0() {
  asm volatile("s_wait_asynccnt 0x0" ::: "memory");
}

// ---------------------------------------------------------------------------
// init: zero sync counters and the 4 hidden-state double buffers
// ---------------------------------------------------------------------------
__global__ void init_ws_kernel(unsigned* __restrict__ counters,
                               float* __restrict__ hzero) {
  int i = blockIdx.x * blockDim.x + threadIdx.x;
  if (i < 8) counters[i] = 0u;
  if (i < 4 * 32 * 512) hzero[i] = 0.0f;
}

// ---------------------------------------------------------------------------
// one-shot fp32 -> bf16 weight conversion (n4 = element_count / 4)
// ---------------------------------------------------------------------------
__global__ __launch_bounds__(256) void f32_to_bf16_kernel(
    const float* __restrict__ in, __bf16* __restrict__ out, int n4) {
  int i = blockIdx.x * 256 + threadIdx.x;
  if (i >= n4) return;
  float4 v = ((const float4*)in)[i];
  __bf16* d = out + (size_t)i * 4;
  d[0] = (__bf16)v.x; d[1] = (__bf16)v.y; d[2] = (__bf16)v.z; d[3] = (__bf16)v.w;
}

// ---------------------------------------------------------------------------
// embedding gather: X[row, :] = bf16(emb[tok[row], :])   (512 cols)
// ---------------------------------------------------------------------------
__global__ __launch_bounds__(128) void embed_gather_kernel(
    const int* __restrict__ tok, const float* __restrict__ emb,
    __bf16* __restrict__ X, int nrows) {
  int row = blockIdx.x;
  if (row >= nrows) return;
  int t = tok[row];
  float4 v = ((const float4*)(emb + (size_t)t * 512))[threadIdx.x];
  __bf16* d = X + (size_t)row * 512 + threadIdx.x * 4;
  d[0] = (__bf16)v.x; d[1] = (__bf16)v.y; d[2] = (__bf16)v.z; d[3] = (__bf16)v.w;
}

// ---------------------------------------------------------------------------
// GEMM: C[M,N] = A[M,K](bf16) * Bw[N,K](bf16)^T + bias[N]
// block tile 32x128, 256 threads = 8 waves, wave = 2 WMMA tiles.
// Tiles staged via global_load_async_to_lds_b128 into ping-pong LDS buffers;
// chunk k+1 copies are in flight underneath chunk k's WMMAs.
// K multiple of 32 (=512 here), M multiple of 32, N multiple of 128.
// ---------------------------------------------------------------------------
__global__ __launch_bounds__(256) void gemm_bf16_wmma_kernel(
    const __bf16* __restrict__ A, const __bf16* __restrict__ Bw,
    const float* __restrict__ bias, float* __restrict__ C,
    int M, int N, int K) {
  __shared__ __bf16 sA[2][32 * 40];    // +8 bf16 pad per row (80B stride)
  __shared__ __bf16 sB[2][128 * 40];

  const int tid  = threadIdx.x;
  const int wave = tid >> 5, lane = tid & 31, lo = lane & 15, hi = lane >> 4;
  const int m0 = blockIdx.y * 32;
  const int n0 = blockIdx.x * 128;
  const int mt = wave & 1;     // 16-row half
  const int ng = wave >> 1;    // 0..3 -> 32 cols (2 tiles)

  // async staging geometry: 16B chunks, 4 per 64B row
  const int arow = tid >> 2, aseg = tid & 3;          // A: threads 0..127
  const unsigned ldsA[2] = { lds_off(&sA[0][0]), lds_off(&sA[1][0]) };
  const unsigned ldsB[2] = { lds_off(&sB[0][0]), lds_off(&sB[1][0]) };

  auto stage = [&](int kc, int p) {
    if (tid < 128) {
      async_to_lds_b128(ldsA[p] + arow * 80 + aseg * 16,
                        (const char*)A + ((size_t)(m0 + arow) * K + kc) * 2 +
                            aseg * 16);
    }
#pragma unroll
    for (int u = 0; u < 2; ++u) {
      int idx = tid + u * 256;               // 0..511
      int row = idx >> 2, seg = idx & 3;
      async_to_lds_b128(ldsB[p] + row * 80 + seg * 16,
                        (const char*)Bw + ((size_t)(n0 + row) * K + kc) * 2 +
                            seg * 16);
    }
  };

  v8f acc0 = {}, acc1 = {};
  int p = 0;
  stage(0, 0);

  for (int kc = 0; kc < K; kc += 32) {
    wait_async0();        // my copies into buf p landed
    __syncthreads();      // everyone's copies landed; prev reads of p^1 done
    if (kc + 32 < K) stage(kc + 32, p ^ 1);

    // A fragment (ISA 16-bit A 16x32 layout)
    const int ar = mt * 16 + lo;
    v8bf al = *(const v8bf*)(&sA[p][ar * 40 + hi * 8]);
    v8bf ah = *(const v8bf*)(&sA[p][ar * 40 + 16 + hi * 8]);
    v16bf afrag = cat16(al, ah);
    {
      const int nr = ng * 32 + lo;
      v8bf bl = *(const v8bf*)(&sB[p][nr * 40 + hi * 8]);
      v8bf bh = *(const v8bf*)(&sB[p][nr * 40 + 16 + hi * 8]);
      acc0 = wmma_bf16(afrag, cat16(bl, bh), acc0);
    }
    {
      const int nr = ng * 32 + 16 + lo;
      v8bf bl = *(const v8bf*)(&sB[p][nr * 40 + hi * 8]);
      v8bf bh = *(const v8bf*)(&sB[p][nr * 40 + 16 + hi * 8]);
      acc1 = wmma_bf16(afrag, cat16(bl, bh), acc1);
    }
    p ^= 1;
  }

  // epilogue (C/D layout: vgpr j -> row j + 8*hi, col = lo)
#pragma unroll
  for (int j = 0; j < 8; ++j) {
    int m  = m0 + mt * 16 + j + hi * 8;
    int n1 = n0 + ng * 32 + lo;
    int n2 = n1 + 16;
    C[(size_t)m * N + n1] = acc0[j] + bias[n1];
    C[(size_t)m * N + n2] = acc1[j] + bias[n2];
  }
}

// ---------------------------------------------------------------------------
// GRU recurrence: 16 cooperative workgroups, each owns 32 H-columns.
// Whh slice (96x512) bf16 RESIDENT in LDS (~100 KB, CDNA5 320KB LDS).
// Per step: stage h -> LDS bf16, gh = h @ Whh_slice^T via WMMA, fused gates,
// h double-buffered in global, per-step grid barrier.
// ---------------------------------------------------------------------------
__global__ __launch_bounds__(128) void gru_rec_wmma_kernel(
    const float* __restrict__ Whh,   // [1536,512] fp32
    const float* __restrict__ bhh,   // [1536]
    const float* __restrict__ gi,    // [steps*32, 1536] = x@Wih^T + bih
    float* __restrict__ hb0,         // [32,512] parity-0 (holds h_init)
    float* __restrict__ hb1,         // [32,512] parity-1
    __bf16* __restrict__ yout,       // [steps*32, 512] bf16 or nullptr
    int steps, unsigned* __restrict__ counter) {
  extern __shared__ __align__(16) char smem[];
  __bf16* sW  = (__bf16*)smem;                         // [96][520]
  __bf16* sH  = (__bf16*)(smem + 96 * 520 * 2);        // [32][520]
  float*  sGh = (float*)(smem + 128 * 520 * 2);        // [32][100]

  const int tid  = threadIdx.x;
  const int wave = tid >> 5, lane = tid & 31, lo = lane & 15, hi = lane >> 4;
  const int wg = blockIdx.x;         // 0..15
  const int s0 = wg * 32;            // this WG's H-column slice
  const unsigned nb = gridDim.x;

  // Load resident Whh slice: rows {g*512 + s0 + r} for g=0..2, r=0..31
  for (int idx = tid; idx < 96 * 512; idx += 128) {
    int rl = idx >> 9, k = idx & 511;
    int g = rl >> 5, r = rl & 31;
    sW[rl * 520 + k] = (__bf16)Whh[(size_t)(g * 512 + s0 + r) * 512 + k];
  }

  const int mt   = wave & 1;   // batch half (rows 0-15 / 16-31)
  const int nsub = wave >> 1;  // 0/1 -> 48 local gate-cols each

  for (int t = 0; t < steps; ++t) {
    const float* hin  = (t & 1) ? hb1 : hb0;
    float*       hout = (t & 1) ? hb0 : hb1;

    for (int idx = tid; idx < 32 * 512; idx += 128) {
      int b = idx >> 9, k = idx & 511;
      sH[b * 520 + k] = (__bf16)hin[idx];
    }
    __syncthreads();

    v8f acc[3]; acc[0] = {}; acc[1] = {}; acc[2] = {};
    const int ar = mt * 16 + lo;
#pragma unroll 4
    for (int kc = 0; kc < 512; kc += 32) {
      v8bf al = *(const v8bf*)(&sH[ar * 520 + kc + hi * 8]);
      v8bf ah = *(const v8bf*)(&sH[ar * 520 + kc + 16 + hi * 8]);
      v16bf afrag = cat16(al, ah);
#pragma unroll
      for (int nt = 0; nt < 3; ++nt) {
        int nr = nsub * 48 + nt * 16 + lo;
        v8bf bl = *(const v8bf*)(&sW[nr * 520 + kc + hi * 8]);
        v8bf bh = *(const v8bf*)(&sW[nr * 520 + kc + 16 + hi * 8]);
        acc[nt] = wmma_bf16(afrag, cat16(bl, bh), acc[nt]);
      }
    }

#pragma unroll
    for (int nt = 0; nt < 3; ++nt) {
      int n = nsub * 48 + nt * 16 + lo;
#pragma unroll
      for (int j = 0; j < 8; ++j) {
        int m = mt * 16 + j + hi * 8;
        sGh[m * 100 + n] = acc[nt][j];
      }
    }
    __syncthreads();

    const float* gi_t = gi + (size_t)t * 32 * 1536;
    for (int idx = tid; idx < 32 * 32; idx += 128) {
      int b = idx >> 5, jj = idx & 31;
      int col = s0 + jj;
      const float* gr = gi_t + (size_t)b * 1536;
      float ghr = sGh[b * 100 + jj]      + bhh[col];
      float ghz = sGh[b * 100 + 32 + jj] + bhh[512 + col];
      float ghn = sGh[b * 100 + 64 + jj] + bhh[1024 + col];
      float r    = sigmoidf_(gr[col] + ghr);
      float z    = sigmoidf_(gr[512 + col] + ghz);
      float cand = tanhf(gr[1024 + col] + r * ghn);
      float hp   = hin[b * 512 + col];
      float hv   = (1.0f - z) * cand + z * hp;
      hout[b * 512 + col] = hv;
      if (yout) yout[((size_t)t * 32 + b) * 512 + col] = (__bf16)hv;
    }

    // per-step grid barrier (monotonic counter, agent scope, s_sleep backoff)
    __syncthreads();
    if (tid == 0) {
      __threadfence();
      __hip_atomic_fetch_add(counter, 1u, __ATOMIC_RELEASE,
                             __HIP_MEMORY_SCOPE_AGENT);
      unsigned target = (unsigned)(t + 1) * nb;
      while (__hip_atomic_load(counter, __ATOMIC_ACQUIRE,
                               __HIP_MEMORY_SCOPE_AGENT) < target)
        __builtin_amdgcn_s_sleep(2);
    }
    __syncthreads();
  }
}

// ---------------------------------------------------------------------------
// host-side orchestration
// ---------------------------------------------------------------------------
extern "C" void kernel_launch(void* const* d_in, const int* in_sizes, int n_in,
                              void* d_out, int out_size, void* d_ws,
                              size_t ws_size, hipStream_t stream) {
  const int*   src      = (const int*)d_in[0];
  const int*   trg      = (const int*)d_in[1];
  const float* enc_emb  = (const float*)d_in[2];
  const float* enc_Wih0 = (const float*)d_in[3];
  const float* enc_Whh0 = (const float*)d_in[4];
  const float* enc_bih0 = (const float*)d_in[5];
  const float* enc_bhh0 = (const float*)d_in[6];
  const float* enc_Wih1 = (const float*)d_in[7];
  const float* enc_Whh1 = (const float*)d_in[8];
  const float* enc_bih1 = (const float*)d_in[9];
  const float* enc_bhh1 = (const float*)d_in[10];
  const float* dec_emb  = (const float*)d_in[11];
  const float* dec_Wih0 = (const float*)d_in[12];
  const float* dec_Whh0 = (const float*)d_in[13];
  const float* dec_bih0 = (const float*)d_in[14];
  const float* dec_bhh0 = (const float*)d_in[15];
  const float* dec_Wih1 = (const float*)d_in[16];
  const float* dec_Whh1 = (const float*)d_in[17];
  const float* dec_bih1 = (const float*)d_in[18];
  const float* dec_bhh1 = (const float*)d_in[19];
  const float* out_W    = (const float*)d_in[20];
  const float* out_b    = (const float*)d_in[21];
  float* out = (float*)d_out;

  // workspace carve (~58 MB)
  char* ws = (char*)d_ws;
  unsigned* counters = (unsigned*)ws;              // 8 uints (pad 256)
  float* hb = (float*)(ws + 256);                  // 4 x [32,512] f32
  float* h0a = hb;
  float* h0b = hb + 16384;
  float* h1a = hb + 32768;
  float* h1b = hb + 49152;
  size_t off = 256 + 4 * 16384 * sizeof(float);    // 262400
  float*  giBuf = (float*)(ws + off);  off += (size_t)2048 * 1536 * 4;
  __bf16* Xa    = (__bf16*)(ws + off); off += (size_t)2048 * 512 * 2;
  __bf16* Ya    = (__bf16*)(ws + off); off += (size_t)2048 * 512 * 2;
  __bf16* Za    = (__bf16*)(ws + off); off += (size_t)2048 * 512 * 2;
  __bf16* wbE0  = (__bf16*)(ws + off); off += (size_t)1536 * 512 * 2;
  __bf16* wbE1  = (__bf16*)(ws + off); off += (size_t)1536 * 512 * 2;
  __bf16* wbD0  = (__bf16*)(ws + off); off += (size_t)1536 * 512 * 2;
  __bf16* wbD1  = (__bf16*)(ws + off); off += (size_t)1536 * 512 * 2;
  __bf16* wbOut = (__bf16*)(ws + off); off += (size_t)32000 * 512 * 2;

  const size_t recShm = (size_t)(96 + 32) * 520 * 2 + (size_t)32 * 100 * 4;

  init_ws_kernel<<<256, 256, 0, stream>>>(counters, hb);

  // one-shot weight conversions (B matrices of all WMMA GEMMs)
  const int wih4 = 1536 * 512 / 4;                 // 196608
  f32_to_bf16_kernel<<<(wih4 + 255) / 256, 256, 0, stream>>>(enc_Wih0, wbE0, wih4);
  f32_to_bf16_kernel<<<(wih4 + 255) / 256, 256, 0, stream>>>(enc_Wih1, wbE1, wih4);
  f32_to_bf16_kernel<<<(wih4 + 255) / 256, 256, 0, stream>>>(dec_Wih0, wbD0, wih4);
  f32_to_bf16_kernel<<<(wih4 + 255) / 256, 256, 0, stream>>>(dec_Wih1, wbD1, wih4);
  const int ow4 = 32000 * 512 / 4;                 // 4096000
  f32_to_bf16_kernel<<<(ow4 + 255) / 256, 256, 0, stream>>>(out_W, wbOut, ow4);

  // ---------------- encoder ----------------
  embed_gather_kernel<<<2048, 128, 0, stream>>>(src, enc_emb, Xa, 2048);

  dim3 gEnc(1536 / 128, 2048 / 32);   // (12, 64)
  gemm_bf16_wmma_kernel<<<gEnc, 256, 0, stream>>>(Xa, wbE0, enc_bih0,
                                                  giBuf, 2048, 1536, 512);
  gru_rec_wmma_kernel<<<16, 128, recShm, stream>>>(enc_Whh0, enc_bhh0, giBuf,
                                                   h0a, h0b, Ya, 64, counters + 0);
  gemm_bf16_wmma_kernel<<<gEnc, 256, 0, stream>>>(Ya, wbE1, enc_bih1,
                                                  giBuf, 2048, 1536, 512);
  gru_rec_wmma_kernel<<<16, 128, recShm, stream>>>(enc_Whh1, enc_bhh1, giBuf,
                                                   h1a, h1b, nullptr, 64, counters + 1);

  // ---------------- decoder (h_init = encoder finals, already in h0a/h1a) ---
  embed_gather_kernel<<<1504, 128, 0, stream>>>(trg, dec_emb, Xa, 1504);

  dim3 gDec(1536 / 128, 1504 / 32);   // (12, 47)
  gemm_bf16_wmma_kernel<<<gDec, 256, 0, stream>>>(Xa, wbD0, dec_bih0,
                                                  giBuf, 1504, 1536, 512);
  gru_rec_wmma_kernel<<<16, 128, recShm, stream>>>(dec_Whh0, dec_bhh0, giBuf,
                                                   h0a, h0b, Ya, 47, counters + 2);
  gemm_bf16_wmma_kernel<<<gDec, 256, 0, stream>>>(Ya, wbD1, dec_bih1,
                                                  giBuf, 1504, 1536, 512);
  gru_rec_wmma_kernel<<<16, 128, recShm, stream>>>(dec_Whh1, dec_bhh1, giBuf,
                                                   h1a, h1b, Za, 47, counters + 3);

  // ---------------- vocab projection (roofline-dominant GEMM) --------------
  dim3 gOut(32000 / 128, 1504 / 32);  // (250, 47)
  gemm_bf16_wmma_kernel<<<gOut, 256, 0, stream>>>(Za, wbOut, out_b,
                                                  out, 1504, 32000, 512);
}
